// ShGaussianKernelConv_65309272703429
// MI455X (gfx1250) — matrix operand order repaired and checked by reading, not compiled, not run on hardware
//
#include <hip/hip_runtime.h>

typedef __attribute__((ext_vector_type(16))) _Float16 v16h;
typedef __attribute__((ext_vector_type(8)))  float    v8f;

#define NB    2
#define NSRC  4096
#define NTGT  2048
#define PP    32
#define CT    512
#define YROWS 48
#define YSTR  516   // padded f32 row stride (8*516 % 64 banks = 32 -> no D-store conflicts)

#define SMEM_Y_BYTES   (YROWS * YSTR * 4)          // 99072
#define SMEM_A_OFF     SMEM_Y_BYTES                // f16 A: 48 x 32
#define SMEM_A_BYTES   (YROWS * PP * 2)            // 3072
#define SMEM_IDX_OFF   (SMEM_A_OFF + SMEM_A_BYTES) // 102144
#define SMEM_BYTES     (SMEM_IDX_OFF + PP * 4)     // 102272

// ------------------------------------------------------------------
// CG tables: 27 dense (2j+1)x(2l+1)x(2J+1) tensors, 3269 floats total
// ------------------------------------------------------------------
__constant__ int T_J [27] = {1,1,1, 2,2,2, 3,3, 1,1,1, 2,2,2,2, 3,3,3, 1,1, 2,2,2, 3,3,3,3};
__constant__ int T_L [27] = {1,1,1, 1,1,1, 1,1, 2,2,2, 2,2,2,2, 2,2,2, 3,3, 3,3,3, 3,3,3,3};
__constant__ int T_JJ[27] = {0,1,2, 1,2,3, 2,3, 1,2,3, 0,1,2,3, 1,2,3, 2,3, 1,2,3, 0,1,2,3};
__constant__ int T_OFS[27] = {0,9,36, 81,126,201, 306,411, 558,603,678, 783,808,883,1008,
                              1183,1288,1463, 1708,1813, 1960,2065,2240, 2485,2534,2681,2926};

struct cpx { double re, im; };

__device__ inline cpx cmul(cpx a, cpx b) {
  return { a.re*b.re - a.im*b.im, a.re*b.im + a.im*b.re };
}

__device__ inline double dfact(int n) {
  double r = 1.0;
  for (int i = 2; i <= n; ++i) r *= (double)i;
  return r;
}

__device__ double su2cg(int j1, int m1, int j2, int m2, int j3, int m3) {
  if (m1 + m2 != m3) return 0.0;
  int vmin = max(max(-j1 + j2 + m3, -j1 + m1), 0);
  int vmax = min(min(j2 + j3 + m1, j3 - j1 + j2), j3 + m3);
  double c = sqrt((2.0*j3 + 1.0)
      * dfact(j3 + j1 - j2) * dfact(j3 - j1 + j2) * dfact(j1 + j2 - j3) / dfact(j1 + j2 + j3 + 1)
      * dfact(j3 + m3) * dfact(j3 - m3)
      / (dfact(j1 - m1) * dfact(j1 + m1) * dfact(j2 - m2) * dfact(j2 + m2)));
  double s = 0.0;
  for (int v = vmin; v <= vmax; ++v) {
    double t = dfact(j2 + j3 + m1 - v) * dfact(j1 - m1 + v)
             / (dfact(v) * dfact(j3 - j1 + j2 - v) * dfact(j3 + m3 - v) * dfact(v + j1 - j2 - m3));
    s += (((v + j2 + m2) & 1) ? -1.0 : 1.0) * t;
  }
  return c * s;
}

// entry [r][c] of the real->complex change-of-basis matrix (incl. (-i)^l phase)
__device__ cpx rtc(int l, int r, int c) {
  const double is2 = 0.70710678118654752440;
  int m = r - l;
  double re = 0.0, im = 0.0;
  if (m < 0) {
    if (c == 2*l - r)      re = is2;   // col l+|m|
    else if (c == r)       im = -is2;  // col l-|m|
  } else if (m == 0) {
    if (c == l)            re = 1.0;
  } else {
    double sg = (m & 1) ? -1.0 : 1.0;
    if (c == r)        re = sg * is2;  // col l+m
    else if (c == 2*l - r) im = sg * is2;  // col l-m
  }
  cpx q = { re, im };
  cpx ph;
  switch (l & 3) {               // (-i)^l
    case 0:  ph = { 1.0,  0.0}; break;
    case 1:  ph = { 0.0, -1.0}; break;
    case 2:  ph = {-1.0,  0.0}; break;
    default: ph = { 0.0,  1.0}; break;
  }
  return cmul(q, ph);
}

__global__ void cg_init_kernel(float* __restrict__ cg) {
  int t = blockIdx.x;                 // one block per table
  int j = T_J[t], l = T_L[t], J = T_JJ[t];
  int dj = 2*j + 1, dl = 2*l + 1, dJ = 2*J + 1;
  int sz = dj * dl * dJ;
  for (int e = threadIdx.x; e < sz; e += blockDim.x) {
    int z = e % dJ;
    int y = (e / dJ) % dl;
    int x = e / (dJ * dl);
    double acc = 0.0;
    for (int i = 0; i < dj; ++i)
      for (int k = 0; k < dl; ++k)
        for (int m = 0; m < dJ; ++m) {
          double cgv = su2cg(j, i - j, l, k - l, J, m - J);
          if (cgv == 0.0) continue;
          cpx qa = rtc(j, i, x);
          cpx qb = rtc(l, k, y);
          cpx qc = rtc(J, m, z); qc.im = -qc.im;   // conj
          cpx p = cmul(cmul(qa, qb), qc);
          acc += p.re * cgv;                        // real part survives by phase choice
        }
    cg[T_OFS[t] + e] = (float)acc;
  }
}

// ------------------------------------------------------------------
// Epilogue block metadata: 34 blocks, reference concat order.
// kind: 0 = l==0 slice, 1 = j==0 slice, 2 = CG contraction
// ------------------------------------------------------------------
struct EpBlk { int J, kind, j, l, ofs, coff; };
constexpr EpBlk EB[34] = {
  // J = 0 (CJ = 384)
  {0,0,0,0,   0,   0}, {0,2,1,1,   0,  96}, {0,2,2,2, 783, 192}, {0,2,3,3,2485, 288},
  // J = 1 (CJ = 864)
  {1,0,0,0,   0,   0}, {1,1,0,1,   0,  96}, {1,2,1,1,   9, 192}, {1,2,2,1,  81, 288},
  {1,2,1,2, 558, 384}, {1,2,2,2, 808, 480}, {1,2,3,2,1183, 576}, {1,2,2,3,1960, 672},
  {1,2,3,3,2534, 768},
  // J = 2 (CJ = 1056)
  {2,0,0,0,   0,   0}, {2,1,0,2,   0,  96}, {2,2,1,1,  36, 192}, {2,2,2,1, 126, 288},
  {2,2,3,1, 306, 384}, {2,2,1,2, 603, 480}, {2,2,2,2, 883, 576}, {2,2,3,2,1288, 672},
  {2,2,1,3,1708, 768}, {2,2,2,3,2065, 864}, {2,2,3,3,2681, 960},
  // J = 3 (CJ = 960)
  {3,0,0,0,   0,   0}, {3,1,0,3,   0,  96}, {3,2,2,1, 201, 192}, {3,2,3,1, 411, 288},
  {3,2,1,2, 678, 384}, {3,2,2,2,1008, 480}, {3,2,3,2,1463, 576}, {3,2,1,3,1813, 672},
  {3,2,2,3,2240, 768}, {3,2,3,3,2926, 864}};

constexpr int      EP_CJ[4]   = {384, 864, 1056, 960};
constexpr int      EP_LOFF[4] = {0, 32, 128, 288};
constexpr unsigned EP_OB[4]   = {0u, 1572864u, 12189696u, 33816576u};

// ------------------------------------------------------------------
// Main kernel: one workgroup (8 wave32 waves) per (b, target)
// ------------------------------------------------------------------
__global__ __launch_bounds__(256)
void sh_conv_kernel(const float* __restrict__ f0, const float* __restrict__ f1,
                    const float* __restrict__ f2, const float* __restrict__ f3,
                    const float* __restrict__ kern, const int* __restrict__ pidx,
                    const float* __restrict__ cg, float* __restrict__ out) {
  extern __shared__ char smem[];
  float*    Y    = (float*)smem;                      // [48][516] f32
  _Float16* Ash  = (_Float16*)(smem + SMEM_A_OFF);    // [48][32]  f16
  int*      idxs = (int*)(smem + SMEM_IDX_OFF);       // [32]

  const int tgt = blockIdx.x;                         // b*NTGT + v
  const int b   = tgt >> 11;

  // Stage A = K^T in f16:  Ash[y=sh*3+s][p] = kernels[b,v,p,sh,s]
  const size_t kbase = (size_t)tgt * (PP * YROWS);
  for (int e = threadIdx.x; e < PP * YROWS; e += blockDim.x) {
    int p = e / YROWS, q = e % YROWS;
    Ash[q * PP + p] = (_Float16)kern[kbase + e];
  }
  if (threadIdx.x < PP) idxs[threadIdx.x] = pidx[(size_t)tgt * PP + threadIdx.x];
  __syncthreads();

  const int wave = threadIdx.x >> 5;
  const int lane = threadIdx.x & 31;
  const int g    = lane >> 4;        // lane group (0: lanes 0-15, 1: 16-31)
  const int ln   = lane & 15;

  // A fragments: 16x32 A tile, contiguous k order shared with B (K = h+16g)
  v16h afr[3];
#pragma unroll
  for (int mt = 0; mt < 3; ++mt) {
    const _Float16* row = &Ash[(mt * 16 + ln) * PP];
#pragma unroll
    for (int h = 0; h < 16; ++h) {
      int k = h + 16 * g;
      afr[mt][h] = row[k];
    }
  }

  // Per-lane gather row indices (K = h + 16g), hoisted out of the tile loop
  int myidx[16];
#pragma unroll
  for (int h = 0; h < 16; ++h) myidx[h] = idxs[h + 16 * g];

  const float* const fb[4] = { f0 + (size_t)b * NSRC * 32,  f1 + (size_t)b * NSRC * 96,
                               f2 + (size_t)b * NSRC * 160, f3 + (size_t)b * NSRC * 224 };

  // Compute Y = A(48x32) x B(32x512), one wmma per 16x16 tile (K = p = 32)
  for (int nt = wave; nt < CT / 16; nt += 8) {
    // segment is wave-uniform: tile boundaries (32/128/288) are multiples of 16
    const int ntu    = __builtin_amdgcn_readfirstlane(nt);
    const int seg    = (ntu >= 2) + (ntu >= 8) + (ntu >= 18);
    const int rowlen = 32 + 64 * seg;                        // {32,96,160,224}
    const int segoff = (seg == 0) ? 0 : (seg == 1) ? 32 : (seg == 2) ? 128 : 288;
    const float* base = (seg == 0) ? fb[0] : (seg == 1) ? fb[1] : (seg == 2) ? fb[2] : fb[3];

    const unsigned cidx = (unsigned)(ntu * 16 + ln - segoff);  // per-lane column in segment

    // Address phase: all 16 gather offsets first
    unsigned off[16];
#pragma unroll
    for (int h = 0; h < 16; ++h)
      off[h] = (unsigned)myidx[h] * (unsigned)rowlen + cidx;
    __builtin_amdgcn_sched_barrier(0);

    // Load phase: 16 independent gathers kept clustered so they pipeline in L2
    float gv[16];
#pragma unroll
    for (int h = 0; h < 16; ++h) gv[h] = base[off[h]];
    __builtin_amdgcn_sched_barrier(0);

    // Convert phase: pack into the B fragment (K = h+16g, N = lane&15)
    v16h bfr;
#pragma unroll
    for (int h = 0; h < 16; ++h) bfr[h] = (_Float16)gv[h];

    const int n = ntu * 16 + ln;
#pragma unroll
    for (int mt = 0; mt < 3; ++mt) {
      v8f acc = {};
      acc = __builtin_amdgcn_wmma_f32_16x16x32_f16(
          /*neg_a=*/false, afr[mt], /*neg_b=*/false, bfr,
          /*c_mod=*/(short)0, acc, /*reuse_a=*/false, /*reuse_b=*/false);
      // D layout: element r -> row M = r + 8g, col N = lane&15
#pragma unroll
      for (int r = 0; r < 8; ++r)
        Y[(mt * 16 + r + 8 * g) * YSTR + n] = acc[r];
    }
  }
  __syncthreads();

  // CG epilogue: uniform fully-unrolled loop over the 34 output blocks
#pragma unroll
  for (int bi = 0; bi < 34; ++bi) {
    const int J    = EB[bi].J;
    const int kind = EB[bi].kind;
    const int j    = EB[bi].j;
    const int l    = EB[bi].l;
    const int ofs  = EB[bi].ofs;
    const int coff = EB[bi].coff;
    const int dJ   = 2 * J + 1;
    const int nel  = dJ * 96;

    for (int t = threadIdx.x; t < nel; t += 256) {
      const int z  = t / 96;
      const int w  = t % 96;
      const int s  = w >> 5;
      const int ch = w & 31;
      float val;
      if (kind == 0) {                       // l==0 slice: sh-block J rows, m=0 column
        val = Y[((J * J + z) * 3 + s) * YSTR + ch];
      } else if (kind == 1) {                // j==0 slice: sh=0 row, degree-l columns
        val = Y[s * YSTR + (EP_LOFF[l] + z * 32 + ch)];
      } else {                               // CG contraction (constant trip counts)
        float a = 0.f;
        for (int x = 0; x < 2 * j + 1; ++x)
          for (int yy = 0; yy < 2 * l + 1; ++yy)
            a += cg[ofs + (x * (2 * l + 1) + yy) * dJ + z] *
                 Y[((j * j + x) * 3 + s) * YSTR + (EP_LOFF[l] + yy * 32 + ch)];
        val = a;
      }
      out[(size_t)EP_OB[J] + ((size_t)tgt * dJ + z) * (size_t)EP_CJ[J] + coff + w] = val;
    }
  }
}

extern "C" void kernel_launch(void* const* d_in, const int* in_sizes, int n_in,
                              void* d_out, int out_size, void* d_ws, size_t ws_size,
                              hipStream_t stream) {
  const float* f0   = (const float*)d_in[0];
  const float* f1   = (const float*)d_in[1];
  const float* f2   = (const float*)d_in[2];
  const float* f3   = (const float*)d_in[3];
  const float* kern = (const float*)d_in[4];
  const int*   pidx = (const int*)d_in[5];
  float* cg  = (float*)d_ws;     // 3269 floats, recomputed every call (deterministic)
  float* out = (float*)d_out;

  cg_init_kernel<<<dim3(27), dim3(256), 0, stream>>>(cg);
  sh_conv_kernel<<<dim3(NB * NTGT), dim3(256), SMEM_BYTES, stream>>>(
      f0, f1, f2, f3, kern, pidx, cg, out);
}
